// MultiHeadAttention_37194416783879
// MI455X (gfx1250) — matrix operand order repaired and checked by reading, not compiled, not run on hardware
//
#include <hip/hip_runtime.h>
#include <hip/hip_bf16.h>

// MI455X / gfx1250 wave32 WMMA multi-head attention forward.
// - All GEMMs: v_wmma_f32_16x16x32_bf16 (bf16 in, f32 accumulate).
// - Scores+softmax fused in 256KB dynamic LDS -> attention written exactly once.
// - K tiles staged into LDS via the Tensor Data Mover (tensor_load_to_lds),
//   double-buffered per wave, synchronized with s_wait_tensorcnt.
// - All fragment loads are 16B vector loads (weights pre-transposed, V stored
//   transposed [B,H,D,S] so every B-fragment is contiguous along K).
//
// Workspace (~34MB of d_ws):
//   [ 0MB) Qh  bf16 [B,NH,S,HD]    8MB
//   [ 8MB) Kh  bf16 [B,NH,S,HD]    8MB
//   [16MB) VhT bf16 [B,NH,HD,S]    8MB
//   [24MB) Ctx bf16 [B,S,HID]      8MB
//   [32MB) WqT,WkT,WvT,WfcT bf16   4 x 0.5MB   (transposed: [out][in])

typedef __attribute__((ext_vector_type(16))) __bf16 v16bf;
typedef __attribute__((ext_vector_type(8)))  __bf16 v8bf;
typedef __attribute__((ext_vector_type(8)))  float  v8f;
typedef __attribute__((ext_vector_type(4)))  float  v4f;
typedef __attribute__((ext_vector_type(4)))  unsigned int v4u;
typedef __attribute__((ext_vector_type(4)))  int    v4i;
typedef __attribute__((ext_vector_type(8)))  int    v8i;

constexpr int B_   = 2;
constexpr int S_   = 4096;
constexpr int HID_ = 512;
constexpr int NH_  = 8;
constexpr int HD_  = 64;
constexpr int MROWS = B_ * S_;        // 8192

__device__ __forceinline__ __bf16 f2bf(float x) { return (__bf16)x; }

__device__ __forceinline__ v16bf cat8(v8bf x, v8bf y) {
  return __builtin_shufflevector(x, y, 0, 1, 2, 3, 4, 5, 6, 7,
                                       8, 9, 10, 11, 12, 13, 14, 15);
}

// ---------------- WMMA fragment loaders (wave32 layouts per CDNA5 ISA) -------
// A fragment 16x32 (MxK) bf16, row-major src, leading dim ld (multiple of 8).
// lane L: row = L&15 ; K = kb..kb+7 and 16+kb..16+kb+7, kb = (L>>4)*8.
__device__ __forceinline__ v16bf load_a_bf16(const __bf16* __restrict__ src, int ld) {
  int lane = threadIdx.x & 31;
  int row  = lane & 15;
  int kb   = (lane >> 4) * 8;
  v8bf lo = *(const v8bf*)(src + row * ld + kb);
  v8bf hi = *(const v8bf*)(src + row * ld + 16 + kb);
  return cat8(lo, hi);
}
// Same but f32 source, converted to bf16 (two float4 loads per half).
__device__ __forceinline__ v16bf load_a_f32(const float* __restrict__ src, int ld) {
  int lane = threadIdx.x & 31;
  int row  = lane & 15;
  int kb   = (lane >> 4) * 8;
  v4f f0 = *(const v4f*)(src + row * ld + kb);
  v4f f1 = *(const v4f*)(src + row * ld + kb + 4);
  v4f f2 = *(const v4f*)(src + row * ld + 16 + kb);
  v4f f3 = *(const v4f*)(src + row * ld + 16 + kb + 4);
  v16bf a;
#pragma unroll
  for (int j = 0; j < 4; ++j) {
    a[j]      = f2bf(f0[j]);
    a[4 + j]  = f2bf(f1[j]);
    a[8 + j]  = f2bf(f2[j]);
    a[12 + j] = f2bf(f3[j]);
  }
  return a;
}
// B fragment 32x16 (KxN) where B[k][n] = src[n*ld + k] (src is N-major / KT).
// lane L: n = L&15 ; K = kb..kb+7 and 16+kb..16+kb+7, kb = (L>>4)*8. Two 16B loads.
__device__ __forceinline__ v16bf load_b_kcontig(const __bf16* __restrict__ src, int ld) {
  int lane = threadIdx.x & 31;
  int n    = lane & 15;
  int kb   = (lane >> 4) * 8;
  v8bf lo = *(const v8bf*)(src + n * ld + kb);
  v8bf hi = *(const v8bf*)(src + n * ld + kb + 16);
  return cat8(lo, hi);
}
// Same pattern but from LDS (emits ds_load_b128 pairs).
__device__ __forceinline__ v16bf load_b_lds(const __bf16* kb16, int dOff) {
  int lane = threadIdx.x & 31;
  int n    = lane & 15;
  int kb   = (lane >> 4) * 8;
  const __bf16* p = kb16 + n * HD_ + dOff + kb;
  v8bf lo = *(const v8bf*)(p);
  v8bf hi = *(const v8bf*)(p + 16);
  return cat8(lo, hi);
}

// ---------------- TDM: 16x64 bf16 tile, global row-major -> LDS row-major ----
// Descriptor per CDNA5 ISA ch.8: group0 {count=1, lds_addr, global_addr, type=2},
// group1 {data_size=2B, tensor_dim0=64, dim0_stride=64, tensor_dim1=16,
//         tile_dim0=64, tile_dim1=16}.
__device__ __forceinline__ void tdm_load_k_tile(unsigned lds_off,
                                                const __bf16* __restrict__ gptr) {
  unsigned long long ga = (unsigned long long)(uintptr_t)gptr;
  v4u g0;
  g0[0] = 1u;                                   // count = 1 valid descriptor
  g0[1] = lds_off;                              // lds_addr (bytes)
  g0[2] = (unsigned)ga;                         // global_addr[31:0]
  g0[3] = (unsigned)(ga >> 32) | (2u << 30);    // global_addr[56:32] | type=2
  v8i g1;
  g1[0] = 0x00010000;                           // data_size = 1 (2 bytes/elem)
  g1[1] = (int)(64u << 16);                     // tensor_dim0[15:0] = 64
  g1[2] = (int)(16u << 16);                     // tensor_dim1[15:0] = 16
  g1[3] = (int)(64u << 16);                     // tile_dim0 = 64
  g1[4] = 16;                                   // tile_dim1 = 16
  g1[5] = 64;                                   // tensor_dim0_stride = 64
  g1[6] = 0;
  g1[7] = 0;
  v4i g2 = {};
  v4i g3 = {};
#if defined(__clang_major__) && (__clang_major__ >= 23)
  v8i g4 = {};
  __builtin_amdgcn_tensor_load_to_lds(g0, g1, g2, g3, g4, 0);
#else
  __builtin_amdgcn_tensor_load_to_lds(g0, g1, g2, g3, 0);
#endif
}

// ---------------- Kernel 0: f32 -> bf16 weight convert + transpose -----------
// dst[n*HID + k] = src[k*HID + n]  (so GEMM B-fragments are K-contiguous)
__global__ void mha_cvt_transpose(const float* __restrict__ src,
                                  __bf16* __restrict__ dst) {
  int k = blockIdx.x * 16 + (threadIdx.x & 15);
  int n = blockIdx.y * 16 + (threadIdx.x >> 4);
  dst[(size_t)n * HID_ + k] = f2bf(src[(size_t)k * HID_ + n]);
}

// ---------------- Kernel 1: QKV projection -----------------------------------
// X [MROWS x HID] f32, Wt [HID(out) x HID(in)] bf16, bias [HID] f32.
// VT=0: Out[b][h][s][d]  (Q,K head-major)
// VT=1: Out[b][h][d][s]  (V transposed so ctx-GEMM B loads are K-contiguous)
template <int VT>
__global__ void mha_proj_gemm(const float* __restrict__ X,
                              const __bf16* __restrict__ Wt,
                              const float* __restrict__ bias,
                              __bf16* __restrict__ Out) {
  int wave = threadIdx.x >> 5;
  int lane = threadIdx.x & 31;
  int m0 = blockIdx.x * 16;
  int n0 = blockIdx.y * 64 + wave * 16;
  v8f c = {};
#pragma unroll
  for (int k0 = 0; k0 < HID_; k0 += 32) {
    v16bf a = load_a_f32(X + (size_t)m0 * HID_ + k0, HID_);
    v16bf b = load_b_kcontig(Wt + (size_t)n0 * HID_ + k0, HID_);
    c = __builtin_amdgcn_wmma_f32_16x16x32_bf16(false, a, false, b, (short)0, c, false, false);
  }
  int n = n0 + (lane & 15);
  int h = n >> 6, d = n & 63;
  float bv = bias[n];
#pragma unroll
  for (int r = 0; r < 8; ++r) {
    int m = m0 + r + ((lane >> 4) << 3);
    int b_ = m >> 12;            // / S_
    int s  = m & (S_ - 1);
    __bf16 val = f2bf(c[r] + bv);
    if (VT)
      Out[(((size_t)b_ * NH_ + h) * HD_ + d) * S_ + s] = val;
    else
      Out[(((size_t)b_ * NH_ + h) * S_ + s) * HD_ + d] = val;
  }
}

// ---------------- Kernel 2: fused scores + softmax ---------------------------
// One block per (bh, 16-query-row tile); 8 waves sweep 256 K-tiles.
// LDS: 16x4096 f32 scores (256KB) + reduce bufs + 8 waves x 2 x 2KB K tiles
// staged by the Tensor Data Mover (double-buffered, s_wait_tensorcnt).
constexpr unsigned SCORE_BYTES = 16u * S_ * 4u;           // 262144
constexpr unsigned RED_BYTES   = (256u + 16u) * 4u;       // 1088
constexpr unsigned KTILE_BYTES = 16u * HD_ * 2u;          // 2048
constexpr unsigned KBUF_BASE   = SCORE_BYTES + RED_BYTES; // 263232 (16B aligned)
constexpr unsigned SMEM_TOTAL  = KBUF_BASE + 8u * 2u * KTILE_BYTES; // 296000

__global__ void mha_scores_softmax(const __bf16* __restrict__ Qh,
                                   const __bf16* __restrict__ Kh,
                                   const int* __restrict__ mask,
                                   float* __restrict__ attn) {
  extern __shared__ char smem_raw[];                  // dynamic LDS (base = 0)
  float* sS    = (float*)smem_raw;                    // [16][S_]
  float* red   = (float*)(smem_raw + SCORE_BYTES);    // [256]
  float* rstat = red + 256;                           // [16]

  int t    = threadIdx.x;
  int wave = t >> 5;
  int lane = t & 31;
  int qt = blockIdx.x;       // 0..255
  int bh = blockIdx.y;       // 0..15
  int b  = bh >> 3;

  unsigned kOff0 = KBUF_BASE + (unsigned)wave * 2u * KTILE_BYTES;
  unsigned kOff1 = kOff0 + KTILE_BYTES;
  __bf16* kbuf0 = (__bf16*)(smem_raw + kOff0);
  __bf16* kbuf1 = (__bf16*)(smem_raw + kOff1);

  const __bf16* Qbase = Qh + ((size_t)bh * S_ + qt * 16) * HD_;
  v16bf a0 = load_a_bf16(Qbase, HD_);        // d = 0..31
  v16bf a1 = load_a_bf16(Qbase + 32, HD_);   // d = 32..63
  const float scale = 0.125f;                // 1/sqrt(HD)

  const __bf16* Ktiles = Kh + (size_t)bh * S_ * HD_;
  // TDM prologue: prime both buffers for this wave's first two tiles.
  tdm_load_k_tile(kOff0, Ktiles + (size_t)(wave) * 16 * HD_);
  tdm_load_k_tile(kOff1, Ktiles + (size_t)(wave + 8) * 16 * HD_);

  int buf = 0;
  for (int kt = wave; kt < S_ / 16; kt += 8) {
    __builtin_amdgcn_s_wait_tensorcnt((short)1);     // oldest TDM (this buf) done
    __bf16* kb16 = buf ? kbuf1 : kbuf0;
    v16bf b0 = load_b_lds(kb16, 0);
    v16bf b1 = load_b_lds(kb16, 32);
    v8f c = {};
    c = __builtin_amdgcn_wmma_f32_16x16x32_bf16(false, a0, false, b0, (short)0, c, false, false);
    c = __builtin_amdgcn_wmma_f32_16x16x32_bf16(false, a1, false, b1, (short)0, c, false, false);
    // refill this buffer with tile kt+16 while the other buffer computes
    if (kt + 16 < S_ / 16)
      tdm_load_k_tile(buf ? kOff1 : kOff0, Ktiles + (size_t)(kt + 16) * 16 * HD_);
    int col = kt * 16 + (lane & 15);
    int mv  = mask[b * S_ + col];
#pragma unroll
    for (int r = 0; r < 8; ++r) {
      int mrow  = r + ((lane >> 4) << 3);
      float val = (mv == 0) ? -1e10f : c[r] * scale;
      sS[mrow * S_ + col] = val;
    }
    buf ^= 1;
  }
  __builtin_amdgcn_s_wait_tensorcnt((short)0);
  __syncthreads();

  // Row softmax: 16 threads per row, 256-element chunks. t = chunk*16 + row.
  int row = t & 15, chunk = t >> 4;
  float* prow = sS + row * S_ + chunk * 256;
  float mx = -3.4e38f;
  for (int i = 0; i < 256; ++i) mx = fmaxf(mx, prow[i]);
  red[t] = mx;
  __syncthreads();
  if (t < 16) {
    float m2 = -3.4e38f;
    for (int cI = 0; cI < 16; ++cI) m2 = fmaxf(m2, red[cI * 16 + t]);
    rstat[t] = m2;
  }
  __syncthreads();
  float rmax = rstat[row];
  float sum = 0.f;
  for (int i = 0; i < 256; ++i) {
    float e = __expf(prow[i] - rmax);
    prow[i] = e;
    sum += e;
  }
  red[t] = sum;
  __syncthreads();
  if (t < 16) {
    float s2 = 0.f;
    for (int cI = 0; cI < 16; ++cI) s2 += red[cI * 16 + t];
    rstat[t] = 1.0f / s2;
  }
  __syncthreads();

  // Single coalesced write of normalized attention.
  float* outbase = attn + ((size_t)bh * S_ + qt * 16) * S_;
  for (int idx = t; idx < 16 * S_; idx += 256) {
    int rr = idx >> 12;            // / S_
    int cc = idx & (S_ - 1);
    outbase[(size_t)rr * S_ + cc] = sS[idx] * rstat[rr];
  }
}

// ---------------- Kernel 3: context = attention @ V --------------------------
// VhT is [B,NH,HD,S] so B[k][n] = VhT[n*S + k] is K-contiguous.
__global__ void mha_ctx_gemm(const float* __restrict__ attn,
                             const __bf16* __restrict__ VhT,
                             __bf16* __restrict__ Ctx) {
  int wave = threadIdx.x >> 5;
  int lane = threadIdx.x & 31;
  int qt = blockIdx.x, bh = blockIdx.y;
  int b = bh >> 3, h = bh & 7;
  const float*  Abase = attn + ((size_t)bh * S_ + qt * 16) * S_;
  const __bf16* Vbase = VhT + (size_t)bh * HD_ * S_ + (size_t)(wave * 16) * S_;
  v8f c = {};
#pragma unroll 4
  for (int k0 = 0; k0 < S_; k0 += 32) {
    if (k0 + 128 < S_)
      __builtin_prefetch(Abase + k0 + 128, 0, 1);
    v16bf a  = load_a_f32(Abase + k0, S_);
    v16bf bb = load_b_kcontig(Vbase + k0, S_);
    c = __builtin_amdgcn_wmma_f32_16x16x32_bf16(false, a, false, bb, (short)0, c, false, false);
  }
  int d = wave * 16 + (lane & 15);
#pragma unroll
  for (int r = 0; r < 8; ++r) {
    int s = qt * 16 + r + ((lane >> 4) << 3);
    Ctx[((size_t)b * S_ + s) * HID_ + h * HD_ + d] = f2bf(c[r]);
  }
}

// ---------------- Kernel 4: output projection (f32 out + bias) ---------------
__global__ void mha_out_gemm(const __bf16* __restrict__ Xbf,
                             const __bf16* __restrict__ Wt,
                             const float* __restrict__ bias,
                             float* __restrict__ Out) {
  int wave = threadIdx.x >> 5;
  int lane = threadIdx.x & 31;
  int m0 = blockIdx.x * 16;
  int n0 = blockIdx.y * 64 + wave * 16;
  v8f c = {};
#pragma unroll
  for (int k0 = 0; k0 < HID_; k0 += 32) {
    v16bf a = load_a_bf16(Xbf + (size_t)m0 * HID_ + k0, HID_);
    v16bf b = load_b_kcontig(Wt + (size_t)n0 * HID_ + k0, HID_);
    c = __builtin_amdgcn_wmma_f32_16x16x32_bf16(false, a, false, b, (short)0, c, false, false);
  }
  int n = n0 + (lane & 15);
  float bv = bias[n];
#pragma unroll
  for (int r = 0; r < 8; ++r) {
    int m = m0 + r + ((lane >> 4) << 3);
    Out[(size_t)m * HID_ + n] = c[r] + bv;
  }
}

// ---------------- Host launch ------------------------------------------------
extern "C" void kernel_launch(void* const* d_in, const int* in_sizes, int n_in,
                              void* d_out, int out_size, void* d_ws, size_t ws_size,
                              hipStream_t stream) {
  const float* q    = (const float*)d_in[0];
  const float* k    = (const float*)d_in[1];
  const float* v    = (const float*)d_in[2];
  const int*   mask = (const int*)d_in[3];
  const float* Wq   = (const float*)d_in[4];
  const float* bq   = (const float*)d_in[5];
  const float* Wk   = (const float*)d_in[6];
  const float* bk   = (const float*)d_in[7];
  const float* Wv   = (const float*)d_in[8];
  const float* bv   = (const float*)d_in[9];
  const float* Wfc  = (const float*)d_in[10];
  const float* bfc  = (const float*)d_in[11];

  float* out  = (float*)d_out;                                   // [B,S,HID]
  float* attn = out + (size_t)B_ * S_ * HID_;                    // [B,NH,S,S]

  char* ws = (char*)d_ws;
  const size_t headElems = (size_t)B_ * NH_ * S_ * HD_;          // 4.19M
  __bf16* wsQ   = (__bf16*)(ws);
  __bf16* wsK   = (__bf16*)(ws + 2 * headElems);
  __bf16* wsVT  = (__bf16*)(ws + 4 * headElems);
  __bf16* wsCtx = (__bf16*)(ws + 6 * headElems);
  __bf16* wWq   = (__bf16*)(ws + 8 * headElems);
  __bf16* wWk   = wWq + (size_t)HID_ * HID_;
  __bf16* wWv   = wWk + (size_t)HID_ * HID_;
  __bf16* wWfc  = wWv + (size_t)HID_ * HID_;

  // Weights -> bf16, transposed to [out][in]
  {
    dim3 g(HID_ / 16, HID_ / 16), blk(256);
    mha_cvt_transpose<<<g, blk, 0, stream>>>(Wq,  wWq);
    mha_cvt_transpose<<<g, blk, 0, stream>>>(Wk,  wWk);
    mha_cvt_transpose<<<g, blk, 0, stream>>>(Wv,  wWv);
    mha_cvt_transpose<<<g, blk, 0, stream>>>(Wfc, wWfc);
  }

  // QKV projections (Q,K head-major; V transposed [B,H,D,S])
  {
    dim3 g(MROWS / 16, HID_ / 64), blk(128);
    mha_proj_gemm<0><<<g, blk, 0, stream>>>(q, wWq, bq, wsQ);
    mha_proj_gemm<0><<<g, blk, 0, stream>>>(k, wWk, bk, wsK);
    mha_proj_gemm<1><<<g, blk, 0, stream>>>(v, wWv, bv, wsVT);
  }

  // Fused scores + softmax (TDM-staged K tiles) -> single 1.07GB attention write
  {
    dim3 g(S_ / 16, B_ * NH_), blk(256);
    mha_scores_softmax<<<g, blk, SMEM_TOTAL, stream>>>(wsQ, wsK, mask, attn);
  }

  // Context GEMM (attention f32 from L2, V transposed bf16)
  {
    dim3 g(S_ / 16, B_ * NH_), blk(128);
    mha_ctx_gemm<<<g, blk, 0, stream>>>(attn, wsVT, wsCtx);
  }

  // Output projection
  {
    dim3 g(MROWS / 16, HID_ / 64), blk(128);
    mha_out_gemm<<<g, blk, 0, stream>>>(wsCtx, wWfc, bfc, out);
  }
}
